// MultiHeadAttention_87660282511561
// MI455X (gfx1250) — matrix operand order repaired and checked by reading, not compiled, not run on hardware
//
#include <hip/hip_runtime.h>
#include <hip/hip_bf16.h>

// MHA: B=4, S=1024, D=1024, H=16, DK=64
// q,k,v fp32 [B,S,D]; causal mask implemented analytically; Wq..Wo fp32 [D,D] (y = x@W.T + b)

typedef __attribute__((ext_vector_type(16))) __bf16 v16bf;
typedef __attribute__((ext_vector_type(8)))  float  v8f;

#define Bb 4
#define Ss 1024
#define Dd 1024
#define Hh 16
#define DK 64

// ---------------------------------------------------------------------------
// GEMM: C[M,N] = A[M,K] * W[N,K]^T + bias ; M=4096, N=1024, K=1024
// block = 256 threads (8 waves), tile 128(M) x 128(N), K-step 32.
// Waves in 4(M) x 2(N): each wave owns 32x64 -> 2 A-frags x 4 B-frags = 8 WMMA/step.
// A_BF16: A already bf16 (async-staged to LDS); else fp32, converted on load.
// OUT_HEADSPLIT: write bf16 to [B,H,S,DK]; else fp32 to [M,N].
// ---------------------------------------------------------------------------
template <bool A_BF16, bool OUT_HEADSPLIT>
__global__ __launch_bounds__(256)
void gemm_wmma(const void* __restrict__ Aptr, const float* __restrict__ W,
               const float* __restrict__ bias, void* __restrict__ Out)
{
    constexpr int Kdim = 1024, Ndim = 1024;
    constexpr int BM = 128, BN = 128, BK = 32;
    __shared__ __align__(16) __bf16 As[BM][BK + 8];
    __shared__ __align__(16) __bf16 Bs[BN][BK + 8];

    const int tid  = threadIdx.x;
    const int wave = tid >> 5, lane = tid & 31;
    const int g = lane >> 4, ml = lane & 15;
    const int wy = wave & 3;          // M sub-tile: wy*32
    const int wx = wave >> 2;         // N sub-tile: wx*64
    const int mtile = blockIdx.x * BM;
    const int ntile = blockIdx.y * BN;

    v8f acc[2][4];
#pragma unroll
    for (int mf = 0; mf < 2; ++mf)
#pragma unroll
        for (int nb = 0; nb < 4; ++nb) acc[mf][nb] = (v8f){0,0,0,0,0,0,0,0};

    const unsigned ldsA = (unsigned)(uintptr_t)&As[0][0];

    for (int k0 = 0; k0 < Kdim; k0 += BK) {
        // ---- stage A tile (128x32) ----
        if constexpr (!A_BF16) {
            const float* A = (const float*)Aptr;
            for (int s = tid; s < BM * BK / 4; s += 256) {     // 1024 float4 slots
                int r = s >> 3, cg = (s & 7) * 4;
                float4 f = *(const float4*)(A + (size_t)(mtile + r) * Kdim + k0 + cg);
                As[r][cg + 0] = (__bf16)f.x; As[r][cg + 1] = (__bf16)f.y;
                As[r][cg + 2] = (__bf16)f.z; As[r][cg + 3] = (__bf16)f.w;
            }
        } else {
            // bf16 A: async DMA global -> LDS (16B per lane), ASYNCcnt-tracked
            const __bf16* A = (const __bf16*)Aptr;
            for (int s = tid; s < BM * BK / 8; s += 256) {     // 512 b128 slots
                int r = s >> 2, cg = (s & 3) * 8;
                unsigned dst = ldsA + (unsigned)(r * (BK + 8) + cg) * 2u;
                const __bf16* src = A + (size_t)(mtile + r) * Kdim + k0 + cg;
                asm volatile("global_load_async_to_lds_b128 %0, %1, off"
                             :: "v"(dst), "v"(src) : "memory");
            }
        }
        // ---- stage W tile (128x32), fp32 -> bf16 ----
        for (int s = tid; s < BN * BK / 4; s += 256) {         // 1024 float4 slots
            int r = s >> 3, cg = (s & 7) * 4;
            float4 f = *(const float4*)(W + (size_t)(ntile + r) * Kdim + k0 + cg);
            Bs[r][cg + 0] = (__bf16)f.x; Bs[r][cg + 1] = (__bf16)f.y;
            Bs[r][cg + 2] = (__bf16)f.z; Bs[r][cg + 3] = (__bf16)f.w;
        }
        if constexpr (A_BF16)
            asm volatile("s_wait_asynccnt 0" ::: "memory");
        __syncthreads();

        // A fragments: row = lane&15 within 16-row frag; halves -> K per ISA layout
        v16bf af[2];
#pragma unroll
        for (int mf = 0; mf < 2; ++mf) {
            const __bf16* row = &As[wy * 32 + mf * 16 + ml][0];
            const int base = 8 * g;
#pragma unroll
            for (int i = 0; i < 8; ++i) af[mf][i]     = row[base + i];
#pragma unroll
            for (int i = 0; i < 8; ++i) af[mf][i + 8] = row[base + 16 + i];
        }
#pragma unroll
        for (int nb = 0; nb < 4; ++nb) {
            v16bf bf;
            const __bf16* rowb = &Bs[wx * 64 + nb * 16 + ml][8 * g];
#pragma unroll
            for (int i = 0; i < 16; ++i) bf[i] = rowb[i + (i & 8)];
#pragma unroll
            for (int mf = 0; mf < 2; ++mf)
                acc[mf][nb] = __builtin_amdgcn_wmma_f32_16x16x32_bf16(
                    false, af[mf], false, bf, (short)0, acc[mf][nb], false, false);
        }
        __syncthreads();
    }

    // ---- writeback ----
#pragma unroll
    for (int mf = 0; mf < 2; ++mf) {
#pragma unroll
        for (int nb = 0; nb < 4; ++nb) {
#pragma unroll
            for (int j = 0; j < 8; ++j) {
                int m = mtile + wy * 32 + mf * 16 + 8 * g + j;
                int n = ntile + wx * 64 + nb * 16 + ml;
                float v = acc[mf][nb][j] + bias[n];
                if constexpr (OUT_HEADSPLIT) {
                    int b = m >> 10, s = m & 1023;     // m = b*S + s
                    int h = n >> 6,  dk = n & 63;      // n = h*DK + dk
                    ((__bf16*)Out)[(((size_t)(b * Hh + h) * Ss + s) * DK) + dk] = (__bf16)v;
                } else {
                    ((float*)Out)[(size_t)m * Ndim + n] = v;
                }
            }
        }
    }
}

// ---------------------------------------------------------------------------
// Causal flash attention over bf16 head-split Q/K/V [B,H,S,DK], DK=64.
// Grid (8 qtiles, H, B), block 256 = 8 waves; wave owns 16 query rows.
// Key tiles of 32 async-staged through LDS; QK^T = 4 wmma, PV = 4 wmma / tile.
// ---------------------------------------------------------------------------
__global__ __launch_bounds__(256)
void attn_wmma(const __bf16* __restrict__ Q, const __bf16* __restrict__ Kh,
               const __bf16* __restrict__ Vh, __bf16* __restrict__ Outp)
{
    __shared__ __align__(16) __bf16 Ks[32][DK];        // 4 KB
    __shared__ __align__(16) __bf16 Vs[32][DK];        // 4 KB
    __shared__ __align__(16) __bf16 Ps[8][16][40];     // per-wave P transpose, 10 KB

    const int tid = threadIdx.x, wave = tid >> 5, lane = tid & 31;
    const int g = lane >> 4, ml = lane & 15;
    const int qt = blockIdx.x, h = blockIdx.y, b = blockIdx.z;
    const size_t headoff = (size_t)(b * Hh + h) * Ss * DK;
    const int q0 = qt * 128 + wave * 16;

    const unsigned ldsK = (unsigned)(uintptr_t)&Ks[0][0] + (unsigned)tid * 16u;
    const unsigned ldsV = (unsigned)(uintptr_t)&Vs[0][0] + (unsigned)tid * 16u;

    // Q fragments, two K=32 chunks spanning DK=64 (registers for whole kernel)
    v16bf aq[2];
    {
        const __bf16* qrow = Q + headoff + (size_t)(q0 + ml) * DK;
#pragma unroll
        for (int c = 0; c < 2; ++c) {
            const int base = c * 32 + 8 * g;
#pragma unroll
            for (int i = 0; i < 8; ++i) aq[c][i]     = qrow[base + i];
#pragma unroll
            for (int i = 0; i < 8; ++i) aq[c][i + 8] = qrow[base + 16 + i];
        }
    }

    v8f o[4];
#pragma unroll
    for (int i = 0; i < 4; ++i) o[i] = (v8f){0,0,0,0,0,0,0,0};
    float mrun[8], lrun[8];
#pragma unroll
    for (int j = 0; j < 8; ++j) { mrun[j] = -3.0e38f; lrun[j] = 0.0f; }

    const int nkt = qt * 4 + 4;                 // causal bound for this q tile
    for (int kt = 0; kt < nkt; ++kt) {
        const int kb = kt * 32;
        // ---- async-stage K,V tiles: 32 keys x 64 bf16 = 16B per thread each ----
        {
            const __bf16* gk = Kh + headoff + (size_t)kb * DK + (size_t)tid * 8;
            const __bf16* gv = Vh + headoff + (size_t)kb * DK + (size_t)tid * 8;
            asm volatile("global_load_async_to_lds_b128 %0, %1, off"
                         :: "v"(ldsK), "v"(gk) : "memory");
            asm volatile("global_load_async_to_lds_b128 %0, %1, off"
                         :: "v"(ldsV), "v"(gv) : "memory");
            asm volatile("s_wait_asynccnt 0" ::: "memory");
        }
        __syncthreads();

        // ---- scores: 16 q-rows x 32 keys (two 16-key groups) ----
        float sc[2][8];
#pragma unroll
        for (int kg = 0; kg < 2; ++kg) {
            v8f a = (v8f){0,0,0,0,0,0,0,0};
#pragma unroll
            for (int c = 0; c < 2; ++c) {          // accumulate over DK
                v16bf bk;
                const __bf16* krow = &Ks[kg * 16 + ml][c * 32 + 8 * g];
#pragma unroll
                for (int i = 0; i < 16; ++i) bk[i] = krow[i + (i & 8)];
                a = __builtin_amdgcn_wmma_f32_16x16x32_bf16(
                    false, aq[c], false, bk, (short)0, a, false, false);
            }
#pragma unroll
            for (int j = 0; j < 8; ++j) {
                int qg = q0 + 8 * g + j;
                int kk = kb + kg * 16 + ml;
                float s = a[j] * 0.125f;           // 1/sqrt(DK)
                sc[kg][j] = (kk <= qg) ? s : -3.0e38f;   // causal mask
            }
        }

        // ---- online softmax (row lives in 16 lanes of a half-wave) ----
        float alpha[8];
#pragma unroll
        for (int j = 0; j < 8; ++j) {
            float mx = fmaxf(sc[0][j], sc[1][j]);
#pragma unroll
            for (int off = 8; off >= 1; off >>= 1)
                mx = fmaxf(mx, __shfl_xor(mx, off, 32));
            float mnew = fmaxf(mrun[j], mx);
            alpha[j] = __expf(mrun[j] - mnew);
            mrun[j] = mnew;
            float p0 = __expf(sc[0][j] - mnew);
            float p1 = __expf(sc[1][j] - mnew);
            sc[0][j] = p0; sc[1][j] = p1;
            float ls = p0 + p1;
#pragma unroll
            for (int off = 8; off >= 1; off >>= 1)
                ls += __shfl_xor(ls, off, 32);
            lrun[j] = lrun[j] * alpha[j] + ls;
        }
#pragma unroll
        for (int d = 0; d < 4; ++d)
#pragma unroll
            for (int j = 0; j < 8; ++j) o[d][j] *= alpha[j];

        // ---- P: C-layout -> A-fragment layout via per-wave LDS ----
#pragma unroll
        for (int kg = 0; kg < 2; ++kg)
#pragma unroll
            for (int j = 0; j < 8; ++j)
                Ps[wave][8 * g + j][kg * 16 + ml] = (__bf16)sc[kg][j];
        asm volatile("s_wait_dscnt 0" ::: "memory");   // wave-internal LDS RAW
        __builtin_amdgcn_wave_barrier();

        v16bf ap;
        {
            const __bf16* prow = &Ps[wave][ml][0];
            const int base = 8 * g;
#pragma unroll
            for (int i = 0; i < 8; ++i) ap[i]     = prow[base + i];
#pragma unroll
            for (int i = 0; i < 8; ++i) ap[i + 8] = prow[base + 16 + i];
        }

        // ---- PV: accumulate O[16 x 64] ----
#pragma unroll
        for (int d = 0; d < 4; ++d) {
            v16bf bv;
#pragma unroll
            for (int i = 0; i < 16; ++i) bv[i] = Vs[8 * g + i + (i & 8)][d * 16 + ml];
            o[d] = __builtin_amdgcn_wmma_f32_16x16x32_bf16(
                false, ap, false, bv, (short)0, o[d], false, false);
        }
        __syncthreads();
    }

    // ---- normalize + store merged-head layout [B,S,D] bf16 ----
#pragma unroll
    for (int d = 0; d < 4; ++d)
#pragma unroll
        for (int j = 0; j < 8; ++j) {
            int q   = q0 + 8 * g + j;
            int col = h * DK + d * 16 + ml;
            Outp[((size_t)b * Ss + q) * Dd + col] = (__bf16)(o[d][j] / lrun[j]);
        }
}

// ---------------------------------------------------------------------------
extern "C" void kernel_launch(void* const* d_in, const int* in_sizes, int n_in,
                              void* d_out, int out_size, void* d_ws, size_t ws_size,
                              hipStream_t stream)
{
    (void)in_sizes; (void)n_in; (void)out_size; (void)ws_size;
    const float* q  = (const float*)d_in[0];
    const float* k  = (const float*)d_in[1];
    const float* v  = (const float*)d_in[2];
    // d_in[3] = causal mask (implemented analytically)
    const float* Wq = (const float*)d_in[4];  const float* bq = (const float*)d_in[5];
    const float* Wk = (const float*)d_in[6];  const float* bk = (const float*)d_in[7];
    const float* Wv = (const float*)d_in[8];  const float* bv = (const float*)d_in[9];
    const float* Wo = (const float*)d_in[10]; const float* bo = (const float*)d_in[11];

    const size_t elems = (size_t)Bb * Ss * Dd;            // 4M elements
    __bf16* qh = (__bf16*)d_ws;
    __bf16* kh = qh + elems;
    __bf16* vh = kh + elems;
    __bf16* at = vh + elems;

    dim3 blk(256, 1, 1);
    dim3 gg(32, 8, 1);                                    // 4096/128 x 1024/128
    gemm_wmma<false, true><<<gg, blk, 0, stream>>>(q, Wq, bq, qh);
    gemm_wmma<false, true><<<gg, blk, 0, stream>>>(k, Wk, bk, kh);
    gemm_wmma<false, true><<<gg, blk, 0, stream>>>(v, Wv, bv, vh);

    attn_wmma<<<dim3(8, Hh, Bb), blk, 0, stream>>>(qh, kh, vh, at);

    gemm_wmma<true, false><<<gg, blk, 0, stream>>>(at, Wo, bo, (float*)d_out);
}
